// CrossAttentionModule_58935541235760
// MI455X (gfx1250) — matrix operand order repaired and checked by reading, not compiled, not run on hardware
//
#include <hip/hip_runtime.h>

namespace {

constexpr int  kB   = 4;
constexpr int  kC   = 2048;
constexpr int  kP   = 1024;                 // H*W
constexpr long kCP  = (long)kC * kP;        // 2,097,152
constexpr long kBCP = (long)kB * kCP;       // 8,388,608
constexpr long kCC  = (long)kC * kC;        // 4,194,304
constexpr long kPP  = (long)kP * kP;        // 1,048,576

typedef __attribute__((ext_vector_type(16))) __bf16          v16bf;
typedef __attribute__((ext_vector_type(8)))  float           v8f;
typedef __attribute__((ext_vector_type(8)))  unsigned short  us8;
typedef __attribute__((ext_vector_type(4)))  unsigned short  us4;

__device__ __forceinline__ unsigned short f2bf(float f) {
  unsigned int u = __float_as_uint(f);
  u += 0x7FFFu + ((u >> 16) & 1u);          // round-to-nearest-even
  return (unsigned short)(u >> 16);
}

// Async copy of 32 contiguous bytes global -> LDS (ASYNCcnt tracked).
// ISA 08_async_tensor §4.4: INST_OFFSET is added to BOTH the LDS and the
// global address, so one (lds,gaddr) pair + offset:16 covers both halves.
__device__ __forceinline__ void async_copy32(const unsigned short* gptr,
                                             unsigned short* lptr) {
  unsigned lds = (unsigned)(reinterpret_cast<uintptr_t>(lptr));
  unsigned long long ga = (unsigned long long)(reinterpret_cast<uintptr_t>(gptr));
  asm volatile("global_load_async_to_lds_b128 %0, %1, off"
               :: "v"(lds), "v"(ga) : "memory");
  asm volatile("global_load_async_to_lds_b128 %0, %1, off offset:16"
               :: "v"(lds), "v"(ga) : "memory");
}

__device__ __forceinline__ void wait_async0() {
  asm volatile("s_wait_asynccnt 0x0" ::: "memory");
}

union FragBF { us8 h[2]; v16bf v; };

// ---------------------------------------------------------------- convert
__global__ void cvt_f32_to_bf16(const float* __restrict__ in,
                                unsigned short* __restrict__ out, int n4) {
  int i = blockIdx.x * blockDim.x + threadIdx.x;
  if (i < n4) {
    float4 v = reinterpret_cast<const float4*>(in)[i];
    us4 o = { f2bf(v.x), f2bf(v.y), f2bf(v.z), f2bf(v.w) };
    reinterpret_cast<us4*>(out)[i] = o;
  }
}

// ---------------------------------------------------------------- GEMM
// C[m,n] = alpha * sum_k A[m,k]*B[k,n]  (+ bias[m])
// TA: A stored in global as (K x M)  -> transpose while staging to LDS
// TB: B stored in global as (K x N)  -> transpose while staging to LDS
//     (!TA/!TB paths stage via GLOBAL_LOAD_ASYNC_TO_LDS_B128)
template <bool TA, bool TB, bool BF16_OUT, bool BIAS>
__global__ __launch_bounds__(256) void
gemm_wmma_bf16(const unsigned short* __restrict__ Ag,
               const unsigned short* __restrict__ Bg,
               float* __restrict__ Cf, unsigned short* __restrict__ C16,
               const float* __restrict__ bias,
               int K, int lda, int ldb, int ldc,
               long sA, long sB, long sC, float alpha) {
  constexpr int BM = 128, BN = 128, BK = 32, LDT = 40;  // LDT%8==0 -> 16B rows
  __shared__ unsigned short As[BM * LDT];
  __shared__ unsigned short Bs[BN * LDT];

  const int tid  = threadIdx.x;
  const int lane = tid & 31;
  const int wave = tid >> 5;          // 8 wave32s
  const int wm   = wave >> 1;         // 0..3 -> 32-row stripes
  const int wn   = wave & 1;          // 0..1 -> 64-col stripes
  const int half = lane >> 4;         // ISA: lanes 16..31 hold the other K/M half
  const int l16  = lane & 15;

  const int m0 = blockIdx.y * BM;
  const int n0 = blockIdx.x * BN;
  const long z = blockIdx.z;
  Ag += z * sA;
  Bg += z * sB;

  v8f acc[2][4];
#pragma unroll
  for (int mi = 0; mi < 2; ++mi)
#pragma unroll
    for (int ni = 0; ni < 4; ++ni) {
      v8f zv = {0.f, 0.f, 0.f, 0.f, 0.f, 0.f, 0.f, 0.f};
      acc[mi][ni] = zv;
    }

  for (int k0 = 0; k0 < K; k0 += BK) {
    bool used_async = false;
    // ---- stage A tile: As[m][k], 128x32 bf16
    if (TA) {
      const int kk = tid >> 3;            // 0..31
      const int mm = (tid & 7) * 16;      // 0..112
      const us8* p = reinterpret_cast<const us8*>(Ag + (long)(k0 + kk) * lda + (m0 + mm));
      us8 r0 = p[0], r1 = p[1];
      if (k0 + BK < K) __builtin_prefetch(Ag + (long)(k0 + BK + kk) * lda + (m0 + mm), 0, 1);
#pragma unroll
      for (int i = 0; i < 8; ++i) As[(mm + i) * LDT + kk] = r0[i];
#pragma unroll
      for (int i = 0; i < 8; ++i) As[(mm + 8 + i) * LDT + kk] = r1[i];
    } else {
      const int mm = tid >> 1;            // 0..127
      const int kk = (tid & 1) * 16;      // 0 / 16
      async_copy32(Ag + (long)(m0 + mm) * lda + (k0 + kk), &As[mm * LDT + kk]);
      used_async = true;
    }
    // ---- stage B tile: Bs[n][k], 128x32 bf16
    if (TB) {
      const int kk = tid >> 3;
      const int nn = (tid & 7) * 16;
      const us8* p = reinterpret_cast<const us8*>(Bg + (long)(k0 + kk) * ldb + (n0 + nn));
      us8 r0 = p[0], r1 = p[1];
      if (k0 + BK < K) __builtin_prefetch(Bg + (long)(k0 + BK + kk) * ldb + (n0 + nn), 0, 1);
#pragma unroll
      for (int i = 0; i < 8; ++i) Bs[(nn + i) * LDT + kk] = r0[i];
#pragma unroll
      for (int i = 0; i < 8; ++i) Bs[(nn + 8 + i) * LDT + kk] = r1[i];
    } else {
      const int nn = tid >> 1;
      const int kk = (tid & 1) * 16;
      async_copy32(Bg + (long)(n0 + nn) * ldb + (k0 + kk), &Bs[nn * LDT + kk]);
      used_async = true;
    }
    if (used_async) wait_async0();   // each wave drains its own async ops
    __syncthreads();                 // then the workgroup tile is coherent

    // ---- load fragments per documented WMMA VGPR layouts
    FragBF fa[2], fb[4];
#pragma unroll
    for (int mi = 0; mi < 2; ++mi) {
      const int mrow = wm * 32 + mi * 16 + l16;          // A: M = lane%16
      const unsigned short* base = &As[mrow * LDT];
      // lanes 0-15: K = 0..7,16..23 ; lanes 16-31: K = 8..15,24..31
      fa[mi].h[0] = *reinterpret_cast<const us8*>(base + half * 8);
      fa[mi].h[1] = *reinterpret_cast<const us8*>(base + 16 + half * 8);
    }
#pragma unroll
    for (int ni = 0; ni < 4; ++ni) {
      const int ncol = wn * 64 + ni * 16 + l16;          // B: N = lane%16
      const unsigned short* base = &Bs[ncol * LDT];
      // lanes 0-15: K = 0..7,16..23 ; lanes 16-31: K = 8..15,24..31
      fb[ni].h[0] = *reinterpret_cast<const us8*>(base + half * 8);
      fb[ni].h[1] = *reinterpret_cast<const us8*>(base + 16 + half * 8);
    }
#pragma unroll
    for (int mi = 0; mi < 2; ++mi)
#pragma unroll
      for (int ni = 0; ni < 4; ++ni)
        acc[mi][ni] = __builtin_amdgcn_wmma_f32_16x16x32_bf16(
            false, fa[mi].v, false, fb[ni].v, (short)0, acc[mi][ni], false, false);
    __syncthreads();
  }

  // ---- epilogue: C/D layout -> VGPR r holds M = r + half*8, N = lane%16
#pragma unroll
  for (int mi = 0; mi < 2; ++mi)
#pragma unroll
    for (int ni = 0; ni < 4; ++ni)
#pragma unroll
      for (int r = 0; r < 8; ++r) {
        const int row = m0 + wm * 32 + mi * 16 + half * 8 + r;
        const int col = n0 + wn * 64 + ni * 16 + l16;
        float v = acc[mi][ni][r] * alpha;
        if (BIAS) v += bias[row];
        if (BF16_OUT)
          C16[z * sC + (long)row * ldc + col] = f2bf(v);
        else
          Cf[z * sC + (long)row * ldc + col] = v;
      }
}

// ---------------------------------------------------------------- softmax
// One 256-thread block per row of 1024; emits bf16 probabilities.
__global__ __launch_bounds__(256) void
softmax_rows_1024(const float* __restrict__ S, unsigned short* __restrict__ A16) {
  __shared__ float red[256];
  const long row = blockIdx.x;
  const float* s = S + row * 1024;
  unsigned short* o = A16 + row * 1024;
  const int t = threadIdx.x;

  float e[4];
  float lmax = -3.0e38f;
#pragma unroll
  for (int i = 0; i < 4; ++i) lmax = fmaxf(lmax, s[t + i * 256]);
  red[t] = lmax;
  __syncthreads();
  for (int st = 128; st > 0; st >>= 1) {
    if (t < st) red[t] = fmaxf(red[t], red[t + st]);
    __syncthreads();
  }
  const float m = red[0];
  __syncthreads();

  float lsum = 0.f;
#pragma unroll
  for (int i = 0; i < 4; ++i) {
    e[i] = __expf(s[t + i * 256] - m);
    lsum += e[i];
  }
  red[t] = lsum;
  __syncthreads();
  for (int st = 128; st > 0; st >>= 1) {
    if (t < st) red[t] += red[t + st];
    __syncthreads();
  }
  const float inv = 1.0f / red[0];
#pragma unroll
  for (int i = 0; i < 4; ++i) o[t + i * 256] = f2bf(e[i] * inv);
}

// ---------------------------------------------------------------- fusion + passthrough copies
// float4-vectorized: fused = fw*rgb + (1-fw)*ind, plus F_rgb/F_indices copies.
__global__ void fuse_and_copy(const float* __restrict__ Frgb,
                              const float* __restrict__ Find,
                              const float* __restrict__ fw_p,
                              float* __restrict__ out) {
  const long i4 = (long)blockIdx.x * blockDim.x + threadIdx.x;   // float4 index
  const long total4 = kBCP / 4;
  if (i4 >= total4) return;
  const float fw = fw_p[0];
  const long idx = i4 * 4;
  const long b = idx / kCP, rem = idx % kCP;
  const float4* maps = reinterpret_cast<const float4*>(out + 3 * kBCP);
  const float4 r = maps[(b * 2 * kCP + rem) / 4];         // rgb_att
  const float4 d = maps[(b * 2 * kCP + kCP + rem) / 4];   // ind_att
  float4 f;
  f.x = fw * r.x + (1.0f - fw) * d.x;
  f.y = fw * r.y + (1.0f - fw) * d.y;
  f.z = fw * r.z + (1.0f - fw) * d.z;
  f.w = fw * r.w + (1.0f - fw) * d.w;
  reinterpret_cast<float4*>(out)[i4] = f;                                   // fused
  reinterpret_cast<float4*>(out + kBCP)[i4] =
      reinterpret_cast<const float4*>(Frgb)[i4];                            // F_rgb
  reinterpret_cast<float4*>(out + 2 * kBCP)[i4] =
      reinterpret_cast<const float4*>(Find)[i4];                            // F_indices
}

}  // namespace

extern "C" void kernel_launch(void* const* d_in, const int* in_sizes, int n_in,
                              void* d_out, int out_size, void* d_ws, size_t ws_size,
                              hipStream_t stream) {
  (void)in_sizes; (void)n_in; (void)out_size; (void)ws_size;

  const float* F_rgb = (const float*)d_in[0];
  const float* F_ind = (const float*)d_in[1];
  const float* W[6];
  const float* Bi[6];
  for (int j = 0; j < 6; ++j) {            // q_rgb, k_rgb, v_rgb, q_ind, k_ind, v_ind
    W[j]  = (const float*)d_in[2 + 2 * j];
    Bi[j] = (const float*)d_in[3 + 2 * j];
  }
  const float* fw = (const float*)d_in[14];
  float* out = (float*)d_out;

  // ---- workspace layout (bf16 stored as raw ushort)
  unsigned short* ws = (unsigned short*)d_ws;
  unsigned short* X16[2] = {ws, ws + kBCP};          // bf16 activations
  unsigned short* W16 = ws + 2 * kBCP;               // 6 bf16 weight matrices
  unsigned short* QKV = W16 + 6 * kCC;               // 6 bf16 (B,C,HW) tensors
  unsigned short* tail = QKV + 6 * kBCP;
  float* S = (float*)tail;                           // f32 scores (B,HW,HW)
  unsigned short* A16 = (unsigned short*)(S + kB * kPP);  // bf16 probabilities

  const dim3 blk(256);
  const float scale = 1.0f / sqrtf((float)kC);

  // 1) f32 -> bf16 conversions
  cvt_f32_to_bf16<<<(int)(kBCP / 4 / 256), blk, 0, stream>>>(F_rgb, X16[0], (int)(kBCP / 4));
  cvt_f32_to_bf16<<<(int)(kBCP / 4 / 256), blk, 0, stream>>>(F_ind, X16[1], (int)(kBCP / 4));
  for (int j = 0; j < 6; ++j)
    cvt_f32_to_bf16<<<(int)(kCC / 4 / 256), blk, 0, stream>>>(W[j], W16 + j * kCC, (int)(kCC / 4));

  // 2) six 1x1-conv GEMMs:  Y[o,p] = sum_c Wj[o,c] X[c,p] + b[o]   (bf16 out)
  {
    const dim3 grid(kP / 128, kC / 128, kB);
    for (int j = 0; j < 6; ++j) {
      const unsigned short* X = X16[j < 3 ? 0 : 1];
      gemm_wmma_bf16<false, true, true, true><<<grid, blk, 0, stream>>>(
          W16 + j * kCC, X, nullptr, QKV + j * kBCP, Bi[j],
          /*K=*/kC, /*lda=*/kC, /*ldb=*/kP, /*ldc=*/kP,
          /*sA=*/0, /*sB=*/kCP, /*sC=*/kCP, 1.0f);
    }
  }

  // 3) two cross-attentions; maps slot s of d_out gets V·softmax(QᵀK·scale)ᵀ
  const int qi[2] = {0, 3}, ki[2] = {4, 1}, vi[2] = {5, 2};
  for (int a = 0; a < 2; ++a) {
    // S[i,j] = scale * sum_c Q[c,i] K[c,j]   (A transposed-in-memory, B K-major)
    {
      const dim3 grid(kP / 128, kP / 128, kB);
      gemm_wmma_bf16<true, true, false, false><<<grid, blk, 0, stream>>>(
          QKV + qi[a] * kBCP, QKV + ki[a] * kBCP, S, nullptr, nullptr,
          /*K=*/kC, /*lda=*/kP, /*ldb=*/kP, /*ldc=*/kP,
          /*sA=*/kCP, /*sB=*/kCP, /*sC=*/kPP, scale);
    }
    softmax_rows_1024<<<(int)(kB * kP), blk, 0, stream>>>(S, A16);
    // O[c,i] = sum_j V[c,j] A[i,j]   -> write f32 into attention_maps slot a
    {
      const dim3 grid(kP / 128, kC / 128, kB);
      gemm_wmma_bf16<false, false, false, false><<<grid, blk, 0, stream>>>(
          QKV + vi[a] * kBCP, A16, out + 3 * kBCP + a * kCP, nullptr, nullptr,
          /*K=*/kP, /*lda=*/kP, /*ldb=*/kP, /*ldc=*/kP,
          /*sA=*/kCP, /*sB=*/kPP, /*sC=*/2 * kCP, 1.0f);
    }
  }

  // 4) fused output + passthrough copies
  fuse_and_copy<<<(int)(kBCP / 4 / 256), blk, 0, stream>>>(F_rgb, F_ind, fw, out);
}
